// ScaledDotAttention_88278757802462
// MI455X (gfx1250) — matrix operand order repaired and checked by reading, not compiled, not run on hardware
//
#include <hip/hip_runtime.h>

// ---------------------------------------------------------------------------
// Masked scaled-dot attention for MI455X (gfx1250, wave32, WMMA).
//
// Pipeline (all on `stream`, scratch in d_ws, 112MB used):
//   1. convert  : Key   f32 -> f16                      ws[0   ..   8MB)
//      convert  : Query f32 -> f16 * 1/sqrt(D)          ws[16MB..  48MB)
//   2. transpose: Value f32 -> V^T f16 [DV][NK]         ws[8MB ..  16MB)
//   3. GEMM1 (WMMA f16, LDS-tiled): S = Qs*K^T (f32)    ws[48MB.. 112MB)
//   4. masked softmax: P = mask*exp(S-max)/denom (f16)  ws[16MB..48MB) (aliases Qf16)
//   5. GEMM2 (WMMA f16, LDS-tiled): out = P*V (via V^T) d_out
//
// softmax(s)*m / sum(softmax(s)*m) == m*e^s / sum(m*e^s): the softmax
// denominator cancels, so one masked softmax is exact vs the reference.
//
// GEMM: block tile 128x128 (8 waves, 32x64 per wave), K-step 64, double-
// buffered LDS with ONE barrier per step; each A/B element crosses L2->WGP
// once per block (~67 flops/byte of L2 traffic).
// ---------------------------------------------------------------------------

#define NQ 8192
#define NK 2048
#define DD 2048
#define DV 2048

typedef __attribute__((ext_vector_type(16))) _Float16 v16h;
typedef __attribute__((ext_vector_type(8)))  _Float16 v8h;
typedef __attribute__((ext_vector_type(4)))  _Float16 v4h;
typedef __attribute__((ext_vector_type(8)))  float    v8f;

// ---------------- f32 -> f16 convert (optionally scaled) -------------------
__global__ void cvt_f32_f16(const float* __restrict__ in, _Float16* __restrict__ out,
                            float scale, int n4) {
    int i = blockIdx.x * blockDim.x + threadIdx.x;
    if (i < n4) {
        float4 v = ((const float4*)in)[i];
        v4h o;
        o[0] = (_Float16)(v.x * scale);
        o[1] = (_Float16)(v.y * scale);
        o[2] = (_Float16)(v.z * scale);
        o[3] = (_Float16)(v.w * scale);
        ((v4h*)out)[i] = o;
    }
}

// ---------------- V (f32, [NK][DV]) -> V^T (f16, [DV][NK]) -----------------
__global__ void transpose_v(const float* __restrict__ V, _Float16* __restrict__ VT) {
    __shared__ _Float16 tile[32][33];
    const int tx = threadIdx.x, ty = threadIdx.y;
    const int d0 = blockIdx.x * 32, k0 = blockIdx.y * 32;
#pragma unroll
    for (int i = 0; i < 4; ++i)
        tile[ty + 8 * i][tx] = (_Float16)V[(size_t)(k0 + ty + 8 * i) * DV + d0 + tx];
    __syncthreads();
#pragma unroll
    for (int i = 0; i < 4; ++i)
        VT[(size_t)(d0 + ty + 8 * i) * NK + k0 + tx] = tile[tx][ty + 8 * i];
}

// ---------------- NT GEMM, f16 in / f32 out, WMMA 16x16x32 -----------------
// C[M][N] = A[M][K] * B[N][K]^T.
// Block: 256 threads = 8 waves; C tile 128x128; wave tile 32(M) x 64(N).
// K-step 64 (two 16x16x32 WMMA K-chunks per staged tile).
#define KSTEP   64
#define LDSROW  72   // 64 data halves + 8 pad (row stride 144B = 9*16B)

template <int KDIM, int NDIM>
__global__ __launch_bounds__(256) void gemm_nt_wmma(const _Float16* __restrict__ A,
                                                    const _Float16* __restrict__ B,
                                                    float* __restrict__ C) {
    __shared__ __align__(32) _Float16 As[2][128][LDSROW];
    __shared__ __align__(32) _Float16 Bs[2][128][LDSROW];

    const int tid  = threadIdx.x;
    const int lane = tid & 31;
    const int wave = tid >> 5;
    const int h    = lane >> 4;     // half-wave select (K-offset swizzle)
    const int ln   = lane & 15;
    const int wm   = wave & 3;      // wave M sub-tile (32 rows each)
    const int wn   = wave >> 2;     // wave N sub-tile (64 cols each)
    const int m_base = blockIdx.y * 128;
    const int n_base = blockIdx.x * 128;

    // Staging: each thread moves one 64B segment of A and one of B per step.
    const int srow = tid >> 1;            // 0..127
    const int scol = (tid & 1) << 5;      // 0 or 32 (halves)
    const _Float16* ga = A + (size_t)(m_base + srow) * KDIM + scol;
    const _Float16* gb = B + (size_t)(n_base + srow) * KDIM + scol;

    const v8f zero = {0.f, 0.f, 0.f, 0.f, 0.f, 0.f, 0.f, 0.f};
    v8f acc[2][4] = {{zero, zero, zero, zero}, {zero, zero, zero, zero}};

    // Prologue: stage K-step 0 into buffer 0.
    {
        _Float16* la = &As[0][srow][scol];
        _Float16* lb = &Bs[0][srow][scol];
#pragma unroll
        for (int j = 0; j < 4; ++j) *(v8h*)(la + 8 * j) = *(const v8h*)(ga + 8 * j);
#pragma unroll
        for (int j = 0; j < 4; ++j) *(v8h*)(lb + 8 * j) = *(const v8h*)(gb + 8 * j);
    }
    __syncthreads();

    const int NSTEP = KDIM / KSTEP;
#pragma unroll 2
    for (int i = 0; i < NSTEP; ++i) {
        const int cur = i & 1;
        const bool more = (i + 1) < NSTEP;

        // Issue next step's global loads first (max latency overlap with WMMAs).
        v8h ra[4], rb[4];
        if (more) {
            const _Float16* gan = ga + (size_t)(i + 1) * KSTEP;
            const _Float16* gbn = gb + (size_t)(i + 1) * KSTEP;
#pragma unroll
            for (int j = 0; j < 4; ++j) ra[j] = *(const v8h*)(gan + 8 * j);
#pragma unroll
            for (int j = 0; j < 4; ++j) rb[j] = *(const v8h*)(gbn + 8 * j);
        }

        // 2 K-chunks x 8 WMMAs from LDS.
#pragma unroll
        for (int kb = 0; kb < KSTEP; kb += 32) {
            v16h afr[2];
#pragma unroll
            for (int s = 0; s < 2; ++s) {
                const _Float16* ap = &As[cur][wm * 32 + s * 16 + ln][kb];
                v8h a0 = *(const v8h*)(ap + 8 * h);
                v8h a1 = *(const v8h*)(ap + 16 + 8 * h);
                afr[s] = __builtin_shufflevector(a0, a1, 0, 1, 2, 3, 4, 5, 6, 7,
                                                         8, 9, 10, 11, 12, 13, 14, 15);
            }
#pragma unroll
            for (int t = 0; t < 4; ++t) {
                const _Float16* bp = &Bs[cur][wn * 64 + t * 16 + ln][kb + 8 * h];
                v8h b0 = *(const v8h*)(bp);
                v8h b1 = *(const v8h*)(bp + 16);
                v16h b = __builtin_shufflevector(b0, b1, 0, 1, 2, 3, 4, 5, 6, 7,
                                                        8, 9, 10, 11, 12, 13, 14, 15);
                acc[0][t] = __builtin_amdgcn_wmma_f32_16x16x32_f16(
                    false, afr[0], false, b, (short)0, acc[0][t], false, false);
                acc[1][t] = __builtin_amdgcn_wmma_f32_16x16x32_f16(
                    false, afr[1], false, b, (short)0, acc[1][t], false, false);
            }
        }

        if (more) {
            // Store into the other buffer (disjoint from this step's reads),
            // then a single barrier publishes it and retires this buffer.
            _Float16* la = &As[cur ^ 1][srow][scol];
            _Float16* lb = &Bs[cur ^ 1][srow][scol];
#pragma unroll
            for (int j = 0; j < 4; ++j) *(v8h*)(la + 8 * j) = ra[j];
#pragma unroll
            for (int j = 0; j < 4; ++j) *(v8h*)(lb + 8 * j) = rb[j];
            __syncthreads();
        }
    }

    // C/D layout: VGPR r -> row (..)+r+8h, col (..)+ln.
#pragma unroll
    for (int s = 0; s < 2; ++s) {
#pragma unroll
        for (int t = 0; t < 4; ++t) {
            float* cp = C + (size_t)(m_base + wm * 32 + s * 16 + 8 * h) * NDIM
                          + n_base + wn * 64 + 16 * t + ln;
#pragma unroll
            for (int r = 0; r < 8; ++r) cp[(size_t)r * NDIM] = acc[s][t][r];
        }
    }
}

// ---------------- masked softmax: P = m*exp(S-max) / sum ------------------
__global__ __launch_bounds__(256) void masked_softmax(const float* __restrict__ S,
                                                      const float* __restrict__ mask,
                                                      _Float16* __restrict__ P) {
    const int q = blockIdx.x;
    const int tid = threadIdx.x;
    const int lane = tid & 31, wid = tid >> 5;
    __shared__ float smax[8], ssum[8];
    const size_t row = (size_t)q * NK;

    float sv[8], mv[8];
    float lmax = -3.0e38f;
#pragma unroll
    for (int i = 0; i < 8; ++i) {
        int k = tid + i * 256;
        sv[i] = S[row + k];
        mv[i] = mask[row + k];
        lmax  = fmaxf(lmax, sv[i]);
    }
#pragma unroll
    for (int o = 16; o > 0; o >>= 1) lmax = fmaxf(lmax, __shfl_xor(lmax, o, 32));
    if (lane == 0) smax[wid] = lmax;
    __syncthreads();
    float rmax = smax[0];
#pragma unroll
    for (int j = 1; j < 8; ++j) rmax = fmaxf(rmax, smax[j]);

    float lsum = 0.f;
#pragma unroll
    for (int i = 0; i < 8; ++i) {
        float p = mv[i] * __expf(sv[i] - rmax);
        sv[i] = p;
        lsum += p;
    }
#pragma unroll
    for (int o = 16; o > 0; o >>= 1) lsum += __shfl_xor(lsum, o, 32);
    if (lane == 0) ssum[wid] = lsum;
    __syncthreads();
    float tot = 0.f;
#pragma unroll
    for (int j = 0; j < 8; ++j) tot += ssum[j];
    const float inv = 1.0f / tot;
#pragma unroll
    for (int i = 0; i < 8; ++i)
        P[row + tid + i * 256] = (_Float16)(sv[i] * inv);
}

// ---------------------------------------------------------------------------
extern "C" void kernel_launch(void* const* d_in, const int* in_sizes, int n_in,
                              void* d_out, int out_size, void* d_ws, size_t ws_size,
                              hipStream_t stream) {
    (void)in_sizes; (void)n_in; (void)out_size; (void)ws_size;
    const float* Q  = (const float*)d_in[0];
    const float* K  = (const float*)d_in[1];
    const float* V  = (const float*)d_in[2];
    const float* Mk = (const float*)d_in[3];
    float* out = (float*)d_out;

    char* ws = (char*)d_ws;
    const size_t MB = 1024ull * 1024ull;
    _Float16* Kh  = (_Float16*)(ws + 0);          //  8 MB: Key  f16 [NK][D]
    _Float16* VTh = (_Float16*)(ws + 8 * MB);     //  8 MB: V^T  f16 [DV][NK]
    _Float16* Qh  = (_Float16*)(ws + 16 * MB);    // 32 MB: Q*scale f16 [NQ][D]
    float*    S   = (float*)   (ws + 48 * MB);    // 64 MB: scores f32 [NQ][NK]
    _Float16* P   = Qh;                           // P aliases Qh (dead after GEMM1)

    const float scale = 1.0f / sqrtf((float)DD);

    cvt_f32_f16<<<(NK * DD) / (256 * 4), 256, 0, stream>>>(K, Kh, 1.0f, (NK * DD) / 4);
    cvt_f32_f16<<<(NQ * DD) / (256 * 4), 256, 0, stream>>>(Q, Qh, scale, (NQ * DD) / 4);
    transpose_v<<<dim3(DV / 32, NK / 32), dim3(32, 8), 0, stream>>>(V, VTh);

    gemm_nt_wmma<DD, NK><<<dim3(NK / 128, NQ / 128), 256, 0, stream>>>(Qh, Kh, S);
    masked_softmax<<<NQ, 256, 0, stream>>>(S, Mk, P);
    gemm_nt_wmma<NK, DV><<<dim3(DV / 128, NQ / 128), 256, 0, stream>>>(P, VTh, out);
}
